// TemporalContrastiveLoss_5918464934112
// MI455X (gfx1250) — compile-verified, matrix operand order checked by reading, not run on hardware
//
#include <hip/hip_runtime.h>

typedef __attribute__((ext_vector_type(16))) _Float16 v16h;
typedef __attribute__((ext_vector_type(8)))  _Float16 v8h;
typedef __attribute__((ext_vector_type(8)))  float    v8f;

#define BATCH 8192
#define DIM   512
#define INV_T 10.0f   // 1 / temperature
#define LDSS  40      // padded LDS row stride in f16 elements (80B, 16B-aligned)

// ---------------- init: zero rowsum / colsum ----------------
__global__ __launch_bounds__(256) void tcl_init(float* ws, int n) {
  int i = blockIdx.x * 256 + threadIdx.x;
  if (i < n) ws[i] = 0.0f;
}

// ---------------- diag: fp32-exact diagonal logits ----------------
__global__ __launch_bounds__(256) void tcl_diag(const float* __restrict__ P,
                                                const float* __restrict__ Q,
                                                float* __restrict__ diag) {
  const int row  = blockIdx.x * 8 + (threadIdx.x >> 5);
  const int lane = threadIdx.x & 31;
  const float4* p4 = (const float4*)(P + (size_t)row * DIM);
  const float4* q4 = (const float4*)(Q + (size_t)row * DIM);
  float s = 0.0f;
  #pragma unroll
  for (int kk = 0; kk < 4; ++kk) {
    float4 p = p4[lane + 32 * kk];
    float4 q = q4[lane + 32 * kk];
    s += p.x * q.x + p.y * q.y + p.z * q.z + p.w * q.w;
  }
  s += __shfl_xor(s, 1);
  s += __shfl_xor(s, 2);
  s += __shfl_xor(s, 4);
  s += __shfl_xor(s, 8);
  s += __shfl_xor(s, 16);
  if (lane == 0) diag[row] = s * INV_T;
}

__device__ __forceinline__ v8h cvt8(const float4& a, const float4& b) {
  v8h h;
  h[0] = (_Float16)a.x; h[1] = (_Float16)a.y; h[2] = (_Float16)a.z; h[3] = (_Float16)a.w;
  h[4] = (_Float16)b.x; h[5] = (_Float16)b.y; h[6] = (_Float16)b.z; h[7] = (_Float16)b.w;
  return h;
}

// Load one 16x32 f16 WMMA operand fragment for this lane from an LDS tile.
// Lane l<16: row=l, K 0..7 + 16..23 ; lane l>=16: row=l-16, K 8..15 + 24..31.
__device__ __forceinline__ v16h ld_frag(const _Float16* base) {
  v8h lo = *(const v8h*)base;          // ds_read_b128
  v8h hi = *(const v8h*)(base + 16);   // ds_read_b128
  return __builtin_shufflevector(lo, hi, 0,1,2,3,4,5,6,7,8,9,10,11,12,13,14,15);
}

// ------------- fused GEMM + exp + row/col partial LSE sums -------------
__global__ __launch_bounds__(256) void tcl_gemm_lse(
    const float* __restrict__ P,      // p_t    [B,D] fp32
    const float* __restrict__ Q,      // p_next [B,D] fp32
    float* __restrict__ rowsum,       // [B] sum_j exp(logit_ij)
    float* __restrict__ colsum)       // [B] sum_i exp(logit_ij)
{
  __shared__ __align__(16) _Float16 At[2][128 * LDSS];
  __shared__ __align__(16) _Float16 Bt[2][128 * LDSS];
  __shared__ float rowLocal[128];
  __shared__ float colLocal[128];

  const int tid   = threadIdx.x;
  const int lane  = tid & 31;
  const int wave  = tid >> 5;
  const int waveM = wave & 3;    // 32-row block within tile
  const int waveN = wave >> 2;   // 64-col block within tile
  const int rowBase = blockIdx.y * 128;
  const int colBase = blockIdx.x * 128;
  const int half = lane >> 4;    // which 16-lane half
  const int l16  = lane & 15;

  if (tid < 128) { rowLocal[tid] = 0.0f; colLocal[tid] = 0.0f; }

  v8f acc[2][4] = {};

  const int lr = tid >> 2;          // 0..63 : row for LDS fill
  const int lc = (tid & 3) * 8;     // 0,8,16,24 : k-group for LDS fill

  const float* gpBase = P + (size_t)(rowBase + lr) * DIM + lc;
  const float* gqBase = Q + (size_t)(colBase + lr) * DIM + lc;

  // ---- register prefetch of first global tile ----
  float4 pr[2][2], qr[2][2];
  #pragma unroll
  for (int pass = 0; pass < 2; ++pass) {
    const float4* gp = (const float4*)(gpBase + (size_t)pass * 64 * DIM);
    const float4* gq = (const float4*)(gqBase + (size_t)pass * 64 * DIM);
    pr[pass][0] = gp[0]; pr[pass][1] = gp[1];
    qr[pass][0] = gq[0]; qr[pass][1] = gq[1];
  }

  int cur = 0;
  for (int kb = 0; kb < DIM; kb += 32) {
    // ---- convert + store prefetched tile into LDS buffer `cur` ----
    #pragma unroll
    for (int pass = 0; pass < 2; ++pass) {
      const int r = lr + pass * 64;
      *(v8h*)&At[cur][r * LDSS + lc] = cvt8(pr[pass][0], pr[pass][1]);
      *(v8h*)&Bt[cur][r * LDSS + lc] = cvt8(qr[pass][0], qr[pass][1]);
    }
    __syncthreads();

    // ---- issue global loads for next tile (in flight during WMMAs) ----
    if (kb + 32 < DIM) {
      #pragma unroll
      for (int pass = 0; pass < 2; ++pass) {
        const float4* gp = (const float4*)(gpBase + (size_t)pass * 64 * DIM + kb + 32);
        const float4* gq = (const float4*)(gqBase + (size_t)pass * 64 * DIM + kb + 32);
        pr[pass][0] = gp[0]; pr[pass][1] = gp[1];
        qr[pass][0] = gq[0]; qr[pass][1] = gq[1];
      }
    }

    // ---- compute: 8 WMMAs on buffer `cur`, distinct operand registers ----
    const _Float16* Ab = &At[cur][0];
    const _Float16* Bb = &Bt[cur][0];
    v16h a0 = ld_frag(&Ab[(waveM * 32      + l16) * LDSS + half * 8]);
    v16h a1 = ld_frag(&Ab[(waveM * 32 + 16 + l16) * LDSS + half * 8]);
    v16h b0 = ld_frag(&Bb[(waveN * 64      + l16) * LDSS + half * 8]);
    v16h b1 = ld_frag(&Bb[(waveN * 64 + 16 + l16) * LDSS + half * 8]);
    v16h b2 = ld_frag(&Bb[(waveN * 64 + 32 + l16) * LDSS + half * 8]);
    v16h b3 = ld_frag(&Bb[(waveN * 64 + 48 + l16) * LDSS + half * 8]);

    acc[0][0] = __builtin_amdgcn_wmma_f32_16x16x32_f16(false, a0, false, b0, (short)0, acc[0][0], false, false);
    acc[1][0] = __builtin_amdgcn_wmma_f32_16x16x32_f16(false, a1, false, b0, (short)0, acc[1][0], false, false);
    acc[0][1] = __builtin_amdgcn_wmma_f32_16x16x32_f16(false, a0, false, b1, (short)0, acc[0][1], false, false);
    acc[1][1] = __builtin_amdgcn_wmma_f32_16x16x32_f16(false, a1, false, b1, (short)0, acc[1][1], false, false);
    acc[0][2] = __builtin_amdgcn_wmma_f32_16x16x32_f16(false, a0, false, b2, (short)0, acc[0][2], false, false);
    acc[1][2] = __builtin_amdgcn_wmma_f32_16x16x32_f16(false, a1, false, b2, (short)0, acc[1][2], false, false);
    acc[0][3] = __builtin_amdgcn_wmma_f32_16x16x32_f16(false, a0, false, b3, (short)0, acc[0][3], false, false);
    acc[1][3] = __builtin_amdgcn_wmma_f32_16x16x32_f16(false, a1, false, b3, (short)0, acc[1][3], false, false);

    cur ^= 1;
  }

  // ---- scale by 1/T, exponentiate in place ----
  #pragma unroll
  for (int m = 0; m < 2; ++m)
    #pragma unroll
    for (int t = 0; t < 4; ++t)
      #pragma unroll
      for (int v = 0; v < 8; ++v)
        acc[m][t][v] = __expf(acc[m][t][v] * INV_T);

  // ---- row partial sums: sum over this wave's 64 cols, butterfly over N ----
  #pragma unroll
  for (int m = 0; m < 2; ++m) {
    v8f ct = acc[m][0] + acc[m][1] + acc[m][2] + acc[m][3];
    #pragma unroll
    for (int v = 0; v < 8; ++v) {
      float r = ct[v];
      r += __shfl_xor(r, 1);
      r += __shfl_xor(r, 2);
      r += __shfl_xor(r, 4);
      r += __shfl_xor(r, 8);
      if (l16 == 0)
        atomicAdd(&rowLocal[waveM * 32 + m * 16 + half * 8 + v], r);
    }
  }

  // ---- col partial sums: sum over this wave's 32 rows ----
  #pragma unroll
  for (int t = 0; t < 4; ++t) {
    float c = 0.0f;
    #pragma unroll
    for (int m = 0; m < 2; ++m)
      #pragma unroll
      for (int v = 0; v < 8; ++v) c += acc[m][t][v];
    c += __shfl_xor(c, 16);
    if (half == 0)
      atomicAdd(&colLocal[waveN * 64 + t * 16 + l16], c);
  }

  __syncthreads();
  if (tid < 128) {
    atomicAdd(&rowsum[rowBase + tid], rowLocal[tid]);
    atomicAdd(&colsum[colBase + tid], colLocal[tid]);
  }
}

// ------------- finalize: loss = mean(0.5*(logRS + logCS) - diag) -------------
__global__ __launch_bounds__(256) void tcl_finalize(
    const float* __restrict__ rowsum, const float* __restrict__ colsum,
    const float* __restrict__ diag, float* __restrict__ out)
{
  __shared__ float red[256];
  float s = 0.0f;
  for (int i = threadIdx.x; i < BATCH; i += 256)
    s += 0.5f * (__logf(rowsum[i]) + __logf(colsum[i])) - diag[i];
  red[threadIdx.x] = s;
  __syncthreads();
  #pragma unroll
  for (int o = 128; o > 0; o >>= 1) {
    if (threadIdx.x < o) red[threadIdx.x] += red[threadIdx.x + o];
    __syncthreads();
  }
  if (threadIdx.x == 0) out[0] = red[0] / (float)BATCH;
}

extern "C" void kernel_launch(void* const* d_in, const int* in_sizes, int n_in,
                              void* d_out, int out_size, void* d_ws, size_t ws_size,
                              hipStream_t stream) {
  (void)in_sizes; (void)n_in; (void)out_size; (void)ws_size;
  const float* p_t    = (const float*)d_in[0];
  const float* p_next = (const float*)d_in[1];
  float* ws     = (float*)d_ws;
  float* rowsum = ws;
  float* colsum = ws + BATCH;
  float* diag   = ws + 2 * BATCH;

  const int ninit = 2 * BATCH;
  tcl_init<<<(ninit + 255) / 256, 256, 0, stream>>>(ws, ninit);
  tcl_diag<<<BATCH / 8, 256, 0, stream>>>(p_t, p_next, diag);

  dim3 grid(BATCH / 128, BATCH / 128);
  tcl_gemm_lse<<<grid, 256, 0, stream>>>(p_t, p_next, rowsum, colsum);

  tcl_finalize<<<1, 256, 0, stream>>>(rowsum, colsum, diag, (float*)d_out);
}
